// proximal_87522843561662
// MI455X (gfx1250) — matrix operand order repaired and checked by reading, not compiled, not run on hardware
//
#include <hip/hip_runtime.h>
#include <hip/hip_bf16.h>

typedef __attribute__((ext_vector_type(16))) _Float16 v16h;
typedef __attribute__((ext_vector_type(8)))  _Float16 v8h;
typedef __attribute__((ext_vector_type(8)))  float    v8f;

#define DINLINE __device__ __attribute__((always_inline)) inline
#define WFENCE() asm volatile("s_wait_dscnt 0" ::: "memory")

#define BSZ   8
#define SLEN  12
#define NCH   2000
#define KSEL  32
#define DM    64
#define NHEAD 4
#define DHEAD 16
#define DFF   256
#define NSEQ  (BSZ * NCH)

// repacked-weight offsets in halves (fragment order: [ntile][kstep][lane][16 halves])
#define REP_WSRC       0L
#define REP_ATTN(l, m) (4096L + ((l) * 4 + (m)) * 4096L)
#define REP_FFN1(e)    (53248L + (e) * 16384L)
#define REP_FFN2(e)    (86016L + (e) * 16384L)
#define REP_HALVES     118784L

// ---------------- per-wave LDS working set (one sequence) ----------------
struct SeqS {
    _Float16 a16[16 * 64];    // WMMA A staging (f16, row-major 16x64)
    _Float16 h16[16 * 256];   // FFN hidden (f16, row-major 16x256)
    float    xf[16 * 64];     // main activation (f32)
    float    mf[16 * 64];     // encoder memory (f32)
    float    qf[16 * 64];
    float    kf[16 * 64];     // also reused as attention output
    float    vf[16 * 64];     // also reused as O-proj output
};

// ---------------- WMMA fragment loaders ----------------
// A fragment (16x32 f16): lane l holds row M=l&15; K-halves per ISA table:
//   a[0..7]  = K = kstep*32 + (l>>4)*8 + 0..7
//   a[8..15] = K = kstep*32 + 16 + (l>>4)*8 + 0..7
DINLINE v16h load_a_frag(const _Float16* sA, int strideA, int kstep, int lane) {
    int m  = lane & 15;
    int kb = kstep * 32 + ((lane >> 4) << 3);
    const _Float16* p0 = sA + m * strideA + kb;
    v8h lo = *(const v8h*)(p0);        // 16B aligned (kb multiple of 8 halves)
    v8h hi = *(const v8h*)(p0 + 16);
    return __builtin_shufflevector(lo, hi, 0, 1, 2, 3, 4, 5, 6, 7,
                                   8, 9, 10, 11, 12, 13, 14, 15);
}

// B fragment: pre-repacked contiguous; lane l reads 16 consecutive halves (32B)
DINLINE v16h load_b_frag(const _Float16* rep, int fragIdx, int lane) {
    return *(const v16h*)(rep + ((long)fragIdx * 32 + lane) * 16);
}

// ---------------- wave GEMM: C[16 x 16*NTILES] = A[16 x 32*KSTEPS] * Wrep ----------------
template <int KSTEPS, int NTILES>
DINLINE void wave_gemm(const _Float16* sA, int strideA, const _Float16* repW,
                       float* sC, int strideC, int lane) {
#pragma unroll
    for (int nt = 0; nt < NTILES; ++nt) {
        v8f acc = {};
#pragma unroll
        for (int ks = 0; ks < KSTEPS; ++ks) {
            v16h a = load_a_frag(sA, strideA, ks, lane);
            v16h b = load_b_frag(repW, nt * KSTEPS + ks, lane);
            acc = __builtin_amdgcn_wmma_f32_16x16x32_f16(false, a, false, b,
                                                         (short)0, acc, false, false);
        }
        int n  = nt * 16 + (lane & 15);
        int mb = (lane >> 4) * 8;
#pragma unroll
        for (int r = 0; r < 8; ++r) sC[(mb + r) * strideC + n] = acc[r];
    }
}

template <int KSTEPS, int NTILES>
DINLINE void wave_gemm_relu_h(const _Float16* sA, int strideA, const _Float16* repW,
                              _Float16* sC, int strideC, int lane) {
#pragma unroll
    for (int nt = 0; nt < NTILES; ++nt) {
        v8f acc = {};
#pragma unroll
        for (int ks = 0; ks < KSTEPS; ++ks) {
            v16h a = load_a_frag(sA, strideA, ks, lane);
            v16h b = load_b_frag(repW, nt * KSTEPS + ks, lane);
            acc = __builtin_amdgcn_wmma_f32_16x16x32_f16(false, a, false, b,
                                                         (short)0, acc, false, false);
        }
        int n  = nt * 16 + (lane & 15);
        int mb = (lane >> 4) * 8;
#pragma unroll
        for (int r = 0; r < 8; ++r) {
            float v = acc[r];
            v = v > 0.f ? v : 0.f;
            sC[(mb + r) * strideC + n] = (_Float16)v;
        }
    }
}

// ---------------- small wave-level helpers ----------------
DINLINE void conv_h(const float* src, _Float16* dst, int nelem, int lane) {
    for (int i = lane; i < nelem; i += 32) dst[i] = (_Float16)src[i];
    WFENCE();
}

DINLINE void copy_f(const float* src, float* dst, int nelem, int lane) {
    for (int i = lane; i < nelem; i += 32) dst[i] = src[i];
    WFENCE();
}

// x = LayerNorm(x + y) over rows 0..11 (dim 64); rows 12..15 zeroed
DINLINE void wave_ln_add(float* x, const float* y, int lane) {
    WFENCE();
    if (lane < 12) {
        const float* xr = x + lane * 64;
        const float* yr = y + lane * 64;
        float s = 0.f, ss = 0.f;
        for (int c = 0; c < 64; ++c) {
            float v = xr[c] + yr[c];
            s += v; ss += v * v;
        }
        float m   = s * (1.f / 64.f);
        float var = ss * (1.f / 64.f) - m * m;
        float r   = rsqrtf(var + 1e-5f);
        for (int c = 0; c < 64; ++c) {
            float v = xr[c] + yr[c];
            x[lane * 64 + c] = (v - m) * r;
        }
    } else if (lane < 16) {
        for (int c = 0; c < 64; ++c) x[lane * 64 + c] = 0.f;
    }
    WFENCE();
}

// MHA core: scores+softmax in registers, then A*V; of may alias kf.
DINLINE void wave_attn(const float* qf, const float* kf, const float* vf,
                       float* of, bool causal, int lane) {
    float p[2][12];
#pragma unroll
    for (int ti = 0; ti < 2; ++ti) {
        int t = lane + ti * 32;
        if (t < 48) {
            int h = t & 3, qr = t >> 2;
            const float* qrow = qf + qr * 64 + h * 16;
            float sc[12];
            float mx = -1e30f;
            for (int k = 0; k < 12; ++k) {
                float s;
                if (causal && k > qr) {
                    s = -1e9f;
                } else {
                    s = 0.f;
                    const float* krow = kf + k * 64 + h * 16;
#pragma unroll
                    for (int d = 0; d < 16; ++d) s += qrow[d] * krow[d];
                    s *= 0.25f;  // 1/sqrt(16)
                }
                sc[k] = s;
                mx = fmaxf(mx, s);
            }
            float sum = 0.f;
            for (int k = 0; k < 12; ++k) { float e = __expf(sc[k] - mx); p[ti][k] = e; sum += e; }
            float inv = 1.f / sum;
            for (int k = 0; k < 12; ++k) p[ti][k] *= inv;
        }
    }
    WFENCE();
#pragma unroll
    for (int ti = 0; ti < 2; ++ti) {
        int t = lane + ti * 32;
        if (t < 48) {
            int h = t & 3, qr = t >> 2;
            for (int d = 0; d < 16; ++d) {
                float o = 0.f;
                for (int k = 0; k < 12; ++k) o += p[ti][k] * vf[k * 64 + h * 16 + d];
                of[qr * 64 + h * 16 + d] = o;
            }
        }
    }
    WFENCE();
}

DINLINE void attn_block(SeqS& S, const _Float16* rep,
                        long wq, long wk, long wv, long wo,
                        bool causal, bool cross, int lane) {
    conv_h(S.xf, S.a16, 1024, lane);
    wave_gemm<2, 4>(S.a16, 64, rep + wq, S.qf, 64, lane);
    WFENCE();
    if (cross) conv_h(S.mf, S.a16, 1024, lane);
    wave_gemm<2, 4>(S.a16, 64, rep + wk, S.kf, 64, lane);
    wave_gemm<2, 4>(S.a16, 64, rep + wv, S.vf, 64, lane);
    WFENCE();
    wave_attn(S.qf, S.kf, S.vf, S.kf, causal, lane);  // O -> kf
    conv_h(S.kf, S.a16, 1024, lane);
    wave_gemm<2, 4>(S.a16, 64, rep + wo, S.vf, 64, lane);  // O-proj -> vf
    wave_ln_add(S.xf, S.vf, lane);
}

DINLINE void ffn_block(SeqS& S, const _Float16* rep, long w1, long w2, int lane) {
    conv_h(S.xf, S.a16, 1024, lane);
    wave_gemm_relu_h<2, 16>(S.a16, 64, rep + w1, S.h16, 256, lane);
    WFENCE();
    wave_gemm<8, 4>(S.h16, 256, rep + w2, S.qf, 64, lane);  // -> qf
    wave_ln_add(S.xf, S.qf, lane);
}

// ---------------- kernel 1: sx / xm / std / tgt ----------------
__global__ void k_prep(const float* __restrict__ x_pr, const float* __restrict__ x_p,
                       float* __restrict__ sx, float* __restrict__ xm,
                       float* __restrict__ stdv, float* __restrict__ tgt) {
    int g = blockIdx.x * blockDim.x + threadIdx.x;
    if (g >= NSEQ) return;
    int b = g / NCH, n = g - b * NCH;
    float v[SLEN];
    float s = 0.f;
#pragma unroll
    for (int l = 0; l < SLEN; ++l) {
        v[l] = x_pr[(b * SLEN + l) * NCH + n];
        s += v[l];
    }
    float m = s * (1.f / SLEN);
    float ss = 0.f;
#pragma unroll
    for (int l = 0; l < SLEN; ++l) {
        float d = v[l] - m;
        sx[g * SLEN + l] = v[l];
        xm[g * SLEN + l] = d;
        ss += d * d;
    }
    stdv[g] = sqrtf(ss);
#pragma unroll
    for (int l = 0; l < SLEN; ++l) {
        float t = 0.f;
        for (int p = 0; p < 4; ++p) t += x_p[((b * 4 + p) * SLEN + l) * NCH + n];
        tgt[g * SLEN + l] = t * 0.25f;
    }
}

// ---------------- kernel 2: weight repack f32 -> f16 B-fragment layout ----------------
__global__ void k_repack(const float* __restrict__ src, int Ksrc, int ldN,
                         int ksteps, _Float16* __restrict__ dst, int total) {
    int t = blockIdx.x * blockDim.x + threadIdx.x;
    if (t >= total) return;
    int j    = t & 15;
    int lane = (t >> 4) & 31;
    int frag = t >> 9;
    int nt   = frag / ksteps;
    int ks   = frag - nt * ksteps;
    int K  = ks * 32 + ((lane >> 4) << 3) + (j & 7) + ((j >> 3) << 4);
    int Nn = nt * 16 + (lane & 15);
    float v = (K < Ksrc) ? src[K * ldN + Nn] : 0.f;
    dst[t] = (_Float16)v;
}

// ---------------- kernel 3: correlation row + top-32 (tie: lower index) ----------------
__global__ void k_topk(const float* __restrict__ xm, const float* __restrict__ stdv,
                       int* __restrict__ idxp) {
    __shared__ float vals[NCH];
    __shared__ float rv[256];
    __shared__ int   ri[256];
    __shared__ float xi[SLEN];
    __shared__ float stdi_sh;

    int g = blockIdx.x;
    int b = g / NCH;
    int tid = threadIdx.x;

    if (tid < SLEN) xi[tid] = xm[g * SLEN + tid];
    if (tid == 0) stdi_sh = stdv[g];
    __syncthreads();
    float si = stdi_sh;

    for (int j = tid; j < NCH; j += 256) {
        const float* xj = xm + (size_t)(b * NCH + j) * SLEN;
        float dot = 0.f;
#pragma unroll
        for (int l = 0; l < SLEN; ++l) dot += xi[l] * xj[l];
        vals[j] = dot / (si * stdv[b * NCH + j] + 1e-8f);
    }
    __syncthreads();

    for (int round = 0; round < KSEL; ++round) {
        float bv = -3.0e38f;
        int   bi = 0x7fffffff;
        for (int j = tid; j < NCH; j += 256) {
            float v = vals[j];
            if (v > bv || (v == bv && j < bi)) { bv = v; bi = j; }
        }
        rv[tid] = bv; ri[tid] = bi;
        __syncthreads();
        for (int s2 = 128; s2 > 0; s2 >>= 1) {
            if (tid < s2) {
                float v = rv[tid + s2]; int j = ri[tid + s2];
                if (v > rv[tid] || (v == rv[tid] && j < ri[tid])) { rv[tid] = v; ri[tid] = j; }
            }
            __syncthreads();
        }
        if (tid == 0) {
            idxp[g * KSEL + round] = ri[0];
            vals[ri[0]] = -3.0e38f;
        }
        __syncthreads();
    }
}

// ---------------- kernel 4: full transformer, one wave per sequence ----------------
__global__ __launch_bounds__(64) void k_main(const float* __restrict__ sx,
                                             const int* __restrict__ idxp,
                                             const float* __restrict__ tgt,
                                             const _Float16* __restrict__ rep,
                                             const float* __restrict__ wtgt,
                                             const float* __restrict__ wout,
                                             float* __restrict__ out) {
    __shared__ SeqS smem[2];   // 2 waves/block, ~60KB static LDS

    int wid  = threadIdx.x >> 5;
    int lane = threadIdx.x & 31;
    SeqS& S  = smem[wid];

    int g = blockIdx.x * 2 + wid;   // grid = NSEQ/2 blocks, 2 waves each
    int b = g / NCH;
    int n = g - b * NCH;

    // ---- build tx (16x64 padded, cols: [sx_row, 32 selected rows], rest 0) ----
    for (int i = lane; i < 1024; i += 32) S.a16[i] = (_Float16)0.f;
    WFENCE();
    for (int c = lane; c < 33; c += 32) {
        int srcRow = (c == 0) ? n : idxp[g * KSEL + (c - 1)];
        const float* sp = sx + (size_t)(b * NCH + srcRow) * SLEN;
        for (int l = 0; l < SLEN; ++l) S.a16[l * 64 + c] = (_Float16)sp[l];
    }
    WFENCE();
    wave_gemm<2, 4>(S.a16, 64, rep + REP_WSRC, S.xf, 64, lane);  // src projection
    WFENCE();

    // ---- encoder ----
    attn_block(S, rep, REP_ATTN(0, 0), REP_ATTN(0, 1), REP_ATTN(0, 2), REP_ATTN(0, 3),
               /*causal=*/false, /*cross=*/false, lane);
    ffn_block(S, rep, REP_FFN1(0), REP_FFN2(0), lane);
    copy_f(S.xf, S.mf, 1024, lane);  // mem

    // ---- decoder init: tgt outer W_tgt ----
    for (int i = lane; i < 1024; i += 32) {
        int r = i >> 6, c = i & 63;
        S.xf[i] = (r < SLEN) ? tgt[g * SLEN + r] * wtgt[c] : 0.f;
    }
    WFENCE();

    attn_block(S, rep, REP_ATTN(1, 0), REP_ATTN(1, 1), REP_ATTN(1, 2), REP_ATTN(1, 3),
               /*causal=*/true, /*cross=*/false, lane);
    attn_block(S, rep, REP_ATTN(2, 0), REP_ATTN(2, 1), REP_ATTN(2, 2), REP_ATTN(2, 3),
               /*causal=*/false, /*cross=*/true, lane);
    ffn_block(S, rep, REP_FFN1(1), REP_FFN2(1), lane);

    // ---- output projection ----
    if (lane < SLEN) {
        float a = 0.f;
        for (int d = 0; d < DM; ++d) a += S.xf[lane * 64 + d] * wout[d];
        out[(size_t)g * SLEN + lane] = a;
    }
}

// ---------------- host side ----------------
extern "C" void kernel_launch(void* const* d_in, const int* in_sizes, int n_in,
                              void* d_out, int out_size, void* d_ws, size_t ws_size,
                              hipStream_t stream) {
    (void)in_sizes; (void)n_in; (void)out_size; (void)ws_size;

    const float* x_pr  = (const float*)d_in[0];
    const float* x_p   = (const float*)d_in[1];
    const float* Wsrc  = (const float*)d_in[2];
    const float* Wtgt  = (const float*)d_in[3];
    const float* Wout  = (const float*)d_in[4];
    const float* attnw = (const float*)d_in[5];
    const float* f1    = (const float*)d_in[6];
    const float* f2    = (const float*)d_in[7];

    char* ws = (char*)d_ws;
    size_t off = 0;
    float* sx   = (float*)(ws + off); off += (size_t)NSEQ * SLEN * 4;   // 768000
    float* xm   = (float*)(ws + off); off += (size_t)NSEQ * SLEN * 4;   // 768000
    float* stdv = (float*)(ws + off); off += (size_t)NSEQ * 4;          // 64000
    float* tgt  = (float*)(ws + off); off += (size_t)NSEQ * SLEN * 4;   // 768000
    int*   idxp = (int*)(ws + off);   off += (size_t)NSEQ * KSEL * 4;   // 2048000
    _Float16* rep = (_Float16*)(ws + off);                              // 237568 bytes

    // 1) weight repack (B-fragment layout, f16)
    auto launch_rep = [&](const float* s, int Ks, int ldN, int kst, int ntl, long hoff) {
        int total = kst * ntl * 512;
        k_repack<<<(total + 255) / 256, 256, 0, stream>>>(s, Ks, ldN, kst, rep + hoff, total);
    };
    launch_rep(Wsrc, 33, 64, 2, 4, REP_WSRC);
    for (int l = 0; l < 3; ++l)
        for (int m = 0; m < 4; ++m)
            launch_rep(attnw + (size_t)(l * 4 + m) * 4096, 64, 64, 2, 4, REP_ATTN(l, m));
    for (int e = 0; e < 2; ++e) launch_rep(f1 + (size_t)e * 16384, 64, 256, 2, 16, REP_FFN1(e));
    for (int e = 0; e < 2; ++e) launch_rep(f2 + (size_t)e * 16384, 256, 64, 8, 4, REP_FFN2(e));

    // 2) sx / xm / std / tgt
    k_prep<<<(NSEQ + 255) / 256, 256, 0, stream>>>(x_pr, x_p, sx, xm, stdv, tgt);

    // 3) correlation + top-32 per row
    k_topk<<<NSEQ, 256, 0, stream>>>(xm, stdv, idxp);

    // 4) transformer, one wave per sequence (2 waves / block)
    k_main<<<NSEQ / 2, 64, 0, stream>>>(sx, idxp, tgt, rep, Wtgt, Wout, (float*)d_out);
}
